// AGCRN_24730421690463
// MI455X (gfx1250) — matrix operand order repaired and checked by reading, not compiled, not run on hardware
//
#include <hip/hip_runtime.h>

#define Nn   4096
#define Bb   32
#define Din  10
#define CIn  2
#define CO   64
#define CiC  66           // CIn + CO
#define COLS (Bb * CiC)   // 2112
#define KTOT 198          // 3 * 66
#define KP   224          // padded K (7 * 32)
#define KS   232          // padded LDS row stride (halfs), 16B-friendly

typedef __attribute__((ext_vector_type(16))) __bf16 v16bf;
typedef __attribute__((ext_vector_type(8)))  float  v8f;

union FragB { uint4 q[2]; v16bf v; };

__device__ __forceinline__ unsigned short f2bf(float f) {
  unsigned u = __float_as_uint(f);
  u += 0x7FFFu + ((u >> 16) & 1u);          // round-to-nearest-even
  return (unsigned short)(u >> 16);
}
__device__ __forceinline__ float bf2f(unsigned short h) {
  return __uint_as_float(((unsigned)h) << 16);
}

__device__ __forceinline__ v8f wmma_bf16(const v16bf& a, const v16bf& b, const v8f& c) {
#if defined(__HIP_DEVICE_COMPILE__)
  return __builtin_amdgcn_wmma_f32_16x16x32_bf16(false, a, false, b, (short)0, c, false, false);
#else
  return c;
#endif
}

// CDNA5 async global->LDS copy (16B per lane), tracked with ASYNCcnt.
__device__ __forceinline__ void async_ld_b128(unsigned lds_off, const void* gaddr) {
#if defined(__HIP_DEVICE_COMPILE__)
  asm volatile("global_load_async_to_lds_b128 %0, %1, off"
               :: "v"(lds_off), "v"(gaddr) : "memory");
#else
  (void)lds_off; (void)gaddr;
#endif
}
__device__ __forceinline__ void wait_async0() {
#if defined(__HIP_DEVICE_COMPILE__)
  asm volatile("s_wait_asynccnt 0" ::: "memory");
#endif
}

// -------------------- K1: P = softmax(relu(E E^T)) rows, bf16 out ----------
__global__ __launch_bounds__(256) void supports_kernel(
    const float* __restrict__ E, unsigned short* __restrict__ P)
{
  __shared__ float sc[Nn];
  __shared__ float red[256];
  __shared__ float El[Din];
  const int n = blockIdx.x, tid = threadIdx.x;
  if (tid < Din) El[tid] = E[n * Din + tid];
  __syncthreads();

  float lmax = 0.0f;  // relu => all scores >= 0
  for (int m = tid; m < Nn; m += 256) {
    float s = 0.f;
#pragma unroll
    for (int d = 0; d < Din; ++d) s += El[d] * E[m * Din + d];
    s = fmaxf(s, 0.0f);
    sc[m] = s;
    lmax = fmaxf(lmax, s);
  }
  red[tid] = lmax; __syncthreads();
  for (int s2 = 128; s2 > 0; s2 >>= 1) {
    if (tid < s2) red[tid] = fmaxf(red[tid], red[tid + s2]);
    __syncthreads();
  }
  const float rmax = red[0];
  __syncthreads();

  float lsum = 0.f;
  for (int m = tid; m < Nn; m += 256) {
    float e = __expf(sc[m] - rmax);
    sc[m] = e;
    lsum += e;
  }
  red[tid] = lsum; __syncthreads();
  for (int s2 = 128; s2 > 0; s2 >>= 1) {
    if (tid < s2) red[tid] += red[tid + s2];
    __syncthreads();
  }
  const float inv = 1.0f / red[0];
  __syncthreads();

  for (int m = tid; m < Nn; m += 256)
    P[(size_t)n * Nn + m] = f2bf(sc[m] * inv);
}

// -------------------- K2: V[n, b*66+c] = concat(X,H) in bf16 ---------------
__global__ __launch_bounds__(256) void build_v(
    const float* __restrict__ X, const float* __restrict__ H,
    unsigned short* __restrict__ V)
{
  size_t idx = (size_t)blockIdx.x * 256 + threadIdx.x;
  if (idx >= (size_t)Nn * COLS) return;
  int n = (int)(idx / COLS), c = (int)(idx % COLS);
  int b = c / CiC, cc = c - b * CiC;
  float v = (cc < CIn) ? X[((size_t)b * Nn + n) * CIn + cc]
                       : H[((size_t)b * Nn + n) * CO + (cc - CIn)];
  V[idx] = f2bf(v);
}

// ---------- K3/K4/K6/K7: Out = bf16(alpha * P@Bm + beta * Vsub) ------------
// Tile: 256 rows x 64 cols, K-step 32, double-buffered LDS.
// 8 waves: wave w owns rows [w*32, w*32+32) (2 A-frags) x 4 col-tiles
// -> 8 wmma per K-step per wave with 12 ds_load_b128.
// A is staged with global_load_async_to_lds_b128 (ASYNCcnt), B is transposed
// through VGPRs and stored with ds_store_b128. One barrier per K-step.
__global__ __launch_bounds__(256) void gemm_prop(
    const unsigned short* __restrict__ P,
    const unsigned short* __restrict__ Bm,
    const unsigned short* __restrict__ Vsub,
    unsigned short* __restrict__ Out,
    float alpha, float beta)
{
  __shared__ __align__(16) unsigned short Al[2][256 * 40]; // [row][k] stride 40
  __shared__ __align__(16) unsigned short Bl[2][64 * 40];  // [col][k] stride 40
  const int mbase = blockIdx.x * 256;
  const int cbase = blockIdx.y * 64;
  const int tid  = threadIdx.x;
  const int warp = tid >> 5;
  const int lane = tid & 31;
  const int lr   = lane & 15;
  const int hi   = lane >> 4;

  const int ar   = tid;              // A staging row (1 row / thread, 64B)
  const int bcol = tid & 63;         // B staging col
  const int bk   = (tid >> 6) << 3;  // B staging k base {0,8,16,24}

  union BReg { unsigned short s[8]; uint4 q; };

  auto stageA = [&](int kb, int p) {   // async global -> LDS, 4 x 16B / thread
    const unsigned short* src = P + (size_t)(mbase + ar) * Nn + kb;
    unsigned lds = (unsigned)(uintptr_t)(&Al[p][ar * 40]);
#pragma unroll
    for (int q = 0; q < 4; ++q)
      async_ld_b128(lds + q * 16, src + q * 8);
  };
  auto loadB = [&](int kb, BReg& r) {  // gather 8 strided bf16 (transpose)
    const unsigned short* bp = Bm + (size_t)(kb + bk) * COLS + cbase + bcol;
#pragma unroll
    for (int j = 0; j < 8; ++j) r.s[j] = bp[(size_t)j * COLS];
  };
  auto storeB = [&](int p, const BReg& r) {
    *(uint4*)(&Bl[p][bcol * 40 + bk]) = r.q;
  };

  // prologue: stage K-tile 0 into buffer 0
  BReg br;
  stageA(0, 0);
  loadB(0, br);
  storeB(0, br);
  wait_async0();
  __syncthreads();

  v8f acc[2][4] = {};
  int p = 0;
  for (int kb = 0; kb < Nn; kb += 32) {
    const bool more = (kb + 32) < Nn;
    if (more) {                     // issue next-tile loads before compute
      stageA(kb + 32, p ^ 1);
      loadB(kb + 32, br);
    }

    // A frags: lanes 0-15 -> K 0..7 & 16..23, lanes 16-31 -> K 8..15 & 24..31
    FragB ua[2];
#pragma unroll
    for (int s = 0; s < 2; ++s) {
      const unsigned short* arow = &Al[p][(warp * 32 + s * 16 + lr) * 40 + hi * 8];
      ua[s].q[0] = *(const uint4*)(arow);
      ua[s].q[1] = *(const uint4*)(arow + 16);
    }
#pragma unroll
    for (int tt = 0; tt < 4; ++tt) {
      // B frag: lanes 0-15 -> K 0..7 & 16..23 at N=lane; lanes 16-31 -> K 8..15 & 24..31
      FragB ub;
      const unsigned short* brow = &Bl[p][(tt * 16 + lr) * 40 + hi * 8];
      ub.q[0] = *(const uint4*)(brow);
      ub.q[1] = *(const uint4*)(brow + 16);
      acc[0][tt] = wmma_bf16(ua[0].v, ub.v, acc[0][tt]);
      acc[1][tt] = wmma_bf16(ua[1].v, ub.v, acc[1][tt]);
    }

    if (more) {
      storeB(p ^ 1, br);            // write other buffer (no conflict w/ reads)
      wait_async0();                // async A writes landed
    }
    __syncthreads();                // single barrier per K-step
    p ^= 1;
  }

  // Epilogue: C layout VGPR j -> (M = hi*8 + j, N = lr)
#pragma unroll
  for (int s = 0; s < 2; ++s) {
    const int rbase = mbase + warp * 32 + s * 16 + hi * 8;
#pragma unroll
    for (int tt = 0; tt < 4; ++tt) {
      const int col = cbase + tt * 16 + lr;
#pragma unroll
      for (int j = 0; j < 8; ++j) {
        const int row = rbase + j;
        float f = alpha * acc[s][tt][j];
        if (Vsub != nullptr) f += beta * bf2f(Vsub[(size_t)row * COLS + col]);
        Out[(size_t)row * COLS + col] = f2bf(f);
      }
    }
  }
}

// -------------------- K5: gate avwgcn output + fuse V_upd ------------------
// One node per block. A = XG [32 x 224] (bf16, LDS), B = W_n^T [o][k] (LDS).
// Two halves of 64 output cols (Z then R). Vu may alias G0v (row-disjoint).
__global__ __launch_bounds__(256) void gate_kernel(
    const float* __restrict__ E,
    const unsigned short* G0v,
    const unsigned short* __restrict__ G1v,
    const unsigned short* __restrict__ G2v,
    const float* __restrict__ Wg,   // [10][198][128]
    const float* __restrict__ bg,   // [10][128]
    const float* __restrict__ X,
    const float* __restrict__ Hst,
    unsigned short* Vu,
    float* __restrict__ Rout)
{
  __shared__ __align__(16) unsigned short Wt[64 * KS]; // [o][k]
  __shared__ __align__(16) unsigned short Am[32 * KS]; // [b][k]
  __shared__ float El[Din];
  const int n = blockIdx.x, tid = threadIdx.x;
  if (tid < Din) El[tid] = E[n * Din + tid];

  // A fill: Am[b][k*66+i] = G_k[n][b*66+i] (already bf16, straight copy)
  for (int idx = tid; idx < Bb * KTOT; idx += 256) {
    int b = idx / KTOT, r = idx - b * KTOT;
    int k = r / CiC, i = r - k * CiC;
    const unsigned short* src = (k == 0) ? G0v : ((k == 1) ? G1v : G2v);
    Am[b * KS + r] = src[(size_t)n * COLS + b * CiC + i];
  }
  for (int idx = tid; idx < Bb * (KP - KTOT); idx += 256) {
    int b = idx / (KP - KTOT);
    Am[b * KS + KTOT + (idx - b * (KP - KTOT))] = 0;
  }
  __syncthreads();

  const int warp = tid >> 5, lane = tid & 31, lr = lane & 15, hi = lane >> 4;
  const int ct = warp & 3, rh = warp >> 2;

  for (int h = 0; h < 2; ++h) {           // h=0: Z cols, h=1: R cols
    // W_n^T fill (recompute from L2-resident 1MB pool): 10-MAC dot per elem
    for (int idx = tid; idx < KTOT * 64; idx += 256) {
      int kk = idx >> 6, o = idx & 63;
      float s = 0.f;
#pragma unroll
      for (int d = 0; d < Din; ++d)
        s += El[d] * Wg[((size_t)d * KTOT + kk) * 128 + h * 64 + o];
      Wt[o * KS + kk] = f2bf(s);
    }
    for (int idx = tid; idx < 64 * (KP - KTOT); idx += 256) {
      int o = idx & 63;
      Wt[o * KS + KTOT + (idx >> 6)] = 0;
    }
    __syncthreads();

    v8f acc = {};
#pragma unroll
    for (int kt = 0; kt < 7; ++kt) {
      const int kb = kt * 32;
      FragB ua, ub;
      const unsigned short* arow = Am + (rh * 16 + lr) * KS + kb + hi * 8;
      ua.q[0] = *(const uint4*)(arow);
      ua.q[1] = *(const uint4*)(arow + 16);
      const unsigned short* brow = Wt + (ct * 16 + lr) * KS + kb + hi * 8;
      ub.q[0] = *(const uint4*)(brow);
      ub.q[1] = *(const uint4*)(brow + 16);
      acc = wmma_bf16(ua.v, ub.v, acc);
    }

    const int o = ct * 16 + lr;
    float bias = 0.f;
#pragma unroll
    for (int d = 0; d < Din; ++d) bias += El[d] * bg[d * 128 + h * 64 + o];

#pragma unroll
    for (int j = 0; j < 8; ++j) {
      const int b = rh * 16 + hi * 8 + j;
      const float x = acc[j] + bias;
      const float s = 1.f / (1.f + __expf(-x));
      const size_t hoff = ((size_t)b * Nn + n) * CO + o;
      if (h == 0) {   // Z -> V_upd feature part: Z * H
        Vu[(size_t)n * COLS + b * CiC + CIn + o] = f2bf(s * Hst[hoff]);
      } else {        // R
        Rout[hoff] = s;
      }
    }
    __syncthreads();
  }

  // X part of V_upd (reads of V row n all happened before first sync)
  if (tid < Bb * CIn) {
    int b = tid >> 1, c = tid & 1;
    Vu[(size_t)n * COLS + b * CiC + c] = f2bf(X[((size_t)b * Nn + n) * CIn + c]);
  }
}

// -------------------- K8: update avwgcn + GRU combine ----------------------
__global__ __launch_bounds__(256) void update_kernel(
    const float* __restrict__ E,
    const unsigned short* __restrict__ G0v,
    const unsigned short* __restrict__ G1v,
    const unsigned short* __restrict__ G2v,
    const float* __restrict__ Wu,   // [10][198][64]
    const float* __restrict__ bu,   // [10][64]
    const float* __restrict__ Hst,
    const float* __restrict__ Rin,
    float* __restrict__ Out)
{
  __shared__ __align__(16) unsigned short Wt[64 * KS];
  __shared__ __align__(16) unsigned short Am[32 * KS];
  __shared__ float El[Din];
  const int n = blockIdx.x, tid = threadIdx.x;
  if (tid < Din) El[tid] = E[n * Din + tid];

  for (int idx = tid; idx < Bb * KTOT; idx += 256) {
    int b = idx / KTOT, r = idx - b * KTOT;
    int k = r / CiC, i = r - k * CiC;
    const unsigned short* src = (k == 0) ? G0v : ((k == 1) ? G1v : G2v);
    Am[b * KS + r] = src[(size_t)n * COLS + b * CiC + i];
  }
  for (int idx = tid; idx < Bb * (KP - KTOT); idx += 256) {
    int b = idx / (KP - KTOT);
    Am[b * KS + KTOT + (idx - b * (KP - KTOT))] = 0;
  }
  __syncthreads();

  for (int idx = tid; idx < KTOT * 64; idx += 256) {
    int kk = idx >> 6, o = idx & 63;
    float s = 0.f;
#pragma unroll
    for (int d = 0; d < Din; ++d)
      s += El[d] * Wu[((size_t)d * KTOT + kk) * 64 + o];
    Wt[o * KS + kk] = f2bf(s);
  }
  for (int idx = tid; idx < 64 * (KP - KTOT); idx += 256) {
    int o = idx & 63;
    Wt[o * KS + KTOT + (idx >> 6)] = 0;
  }
  __syncthreads();

  const int warp = tid >> 5, lane = tid & 31, lr = lane & 15, hi = lane >> 4;
  const int ct = warp & 3, rh = warp >> 2;

  v8f acc = {};
#pragma unroll
  for (int kt = 0; kt < 7; ++kt) {
    const int kb = kt * 32;
    FragB ua, ub;
    const unsigned short* arow = Am + (rh * 16 + lr) * KS + kb + hi * 8;
    ua.q[0] = *(const uint4*)(arow);
    ua.q[1] = *(const uint4*)(arow + 16);
    const unsigned short* brow = Wt + (ct * 16 + lr) * KS + kb + hi * 8;
    ub.q[0] = *(const uint4*)(brow);
    ub.q[1] = *(const uint4*)(brow + 16);
    acc = wmma_bf16(ua.v, ub.v, acc);
  }

  const int o = ct * 16 + lr;
  float bias = 0.f;
#pragma unroll
  for (int d = 0; d < Din; ++d) bias += El[d] * bu[d * 64 + o];

#pragma unroll
  for (int j = 0; j < 8; ++j) {
    const int b = rh * 16 + hi * 8 + j;
    const float hc = tanhf(acc[j] + bias);
    const size_t off = ((size_t)b * Nn + n) * CO + o;
    const float r = Rin[off];
    Out[off] = r * Hst[off] + (1.f - r) * hc;
  }
}

// ---------------------------------------------------------------------------
extern "C" void kernel_launch(void* const* d_in, const int* in_sizes, int n_in,
                              void* d_out, int out_size, void* d_ws, size_t ws_size,
                              hipStream_t stream)
{
  (void)in_sizes; (void)n_in; (void)out_size; (void)ws_size;
  const float* X  = (const float*)d_in[0];
  const float* H  = (const float*)d_in[1];
  const float* E  = (const float*)d_in[2];
  const float* Wg = (const float*)d_in[3];
  const float* bg = (const float*)d_in[4];
  const float* Wu = (const float*)d_in[5];
  const float* bu = (const float*)d_in[6];
  float* Out = (float*)d_out;

  char* ws = (char*)d_ws;
  size_t off = 0;
  auto alloc = [&](size_t bytes) -> void* {
    void* p = ws + off;
    off += (bytes + 255) & ~(size_t)255;
    return p;
  };
  unsigned short* P  = (unsigned short*)alloc((size_t)Nn * Nn * 2);
  unsigned short* V  = (unsigned short*)alloc((size_t)Nn * COLS * 2);
  unsigned short* G1 = (unsigned short*)alloc((size_t)Nn * COLS * 2);
  unsigned short* G2 = (unsigned short*)alloc((size_t)Nn * COLS * 2);
  float*          R  = (float*)alloc((size_t)Bb * Nn * CO * 4);

  supports_kernel<<<Nn, 256, 0, stream>>>(E, P);

  const size_t tot = (size_t)Nn * COLS;
  build_v<<<(unsigned)((tot + 255) / 256), 256, 0, stream>>>(X, H, V);

  dim3 g(Nn / 256, COLS / 64);
  gemm_prop<<<g, 256, 0, stream>>>(P, V,  nullptr, G1, 1.f,  0.f); // G1 = P V
  gemm_prop<<<g, 256, 0, stream>>>(P, G1, V,       G2, 2.f, -1.f); // G2 = 2P G1 - V

  gate_kernel<<<Nn, 256, 0, stream>>>(E, V, G1, G2, Wg, bg, X, H, V, R); // V <- V_upd

  gemm_prop<<<g, 256, 0, stream>>>(P, V,  nullptr, G1, 1.f,  0.f);
  gemm_prop<<<g, 256, 0, stream>>>(P, G1, V,       G2, 2.f, -1.f);

  update_kernel<<<Nn, 256, 0, stream>>>(E, V, G1, G2, Wu, bu, H, R, Out);
}
